// GraphUnet_38603166057129
// MI455X (gfx1250) — matrix operand order repaired and checked by reading, not compiled, not run on hardware
//
#include <hip/hip_runtime.h>
#include <hip/hip_bf16.h>

typedef __attribute__((ext_vector_type(16))) _Float16 v16h;
typedef __attribute__((ext_vector_type(8)))  _Float16 v8h;
typedef __attribute__((ext_vector_type(4)))  _Float16 v4h;
typedef __attribute__((ext_vector_type(8)))  float    v8f;
typedef __attribute__((ext_vector_type(4)))  float    v4f;

#define LDA 40  // padded LDS row stride in halves (80B rows -> 16B aligned sub-chunks)

__device__ __forceinline__ v16h combine16(v8h lo, v8h hi) {
  return __builtin_shufflevector(lo, hi, 0,1,2,3,4,5,6,7,8,9,10,11,12,13,14,15);
}

// A fragment (16x32 f16), rows [mbase..mbase+15], from LDS tile As[rows][LDA].
// Lanes 0-15: K 0..7 & 16..23 ; lanes 16-31: K 8..15 & 24..31 (ISA 16-bit A layout).
__device__ __forceinline__ v16h load_afrag(const _Float16* As, int mbase, int lane) {
  int m = mbase + (lane & 15);
  int koff = (lane >> 4) * 8;
  v8h lo = *(const v8h*)&As[m * LDA + koff];
  v8h hi = *(const v8h*)&As[m * LDA + 16 + koff];
  return combine16(lo, hi);
}

// B fragment (32x16 f16), cols [nbase..nbase+15], from LDS tile Bs[n][k] (K-contiguous).
// Lanes 0-15: K 0..7 & 16..23 ; lanes 16-31: K 8..15 & 24..31 (matches A layout).
__device__ __forceinline__ v16h load_bfrag(const _Float16* Bs, int nbase, int lane) {
  int n = nbase + (lane & 15);
  int kb = (lane >> 4) * 8;
  v8h lo = *(const v8h*)&Bs[n * LDA + kb];
  v8h hi = *(const v8h*)&Bs[n * LDA + kb + 16];
  return combine16(lo, hi);
}

// ---------------------------------------------------------------------------
// WMMA GEMM: C[M,N] = act(A[M,K] @ op(B) + bias[N])
// TRANSB=false: B is [K,N] row-major.  TRANSB=true: C = A @ B^T with B [N,K].
// Block = 128 threads (4 wave32), 64x128 C tile, 32x64 per wave, K step 32.
// Global tile loads are register-pipelined (issued during the WMMA burst).
// Requires M%64==0, N%128==0, K%32==0 (all call sites satisfy this).
// ---------------------------------------------------------------------------
template <bool TRANSB, bool RELU>
__global__ void __launch_bounds__(128)
gemm_wmma(const float* __restrict__ A, const float* __restrict__ B,
          const float* __restrict__ bias, float* __restrict__ C,
          int M, int N, int K) {
  __shared__ _Float16 As[64 * LDA];
  __shared__ _Float16 Bs[128 * LDA];
  const int t = threadIdx.x;
  const int lane = t & 31, wave = t >> 5;
  const int bm = blockIdx.y * 64, bn = blockIdx.x * 128;
  const int wm = (wave >> 1) * 32, wn = (wave & 1) * 64;
  v8f acc[2][4] = {};
  v4f aR[4], bR[8];

  auto load_tiles = [&](int kt) {
    for (int e = 0; e < 4; ++e) {            // A: 64x32 = 512 float4 groups
      int g = e * 128 + t, r = g >> 3, c = (g & 7) * 4;
      aR[e] = *(const v4f*)&A[(size_t)(bm + r) * K + kt + c];
    }
    if (TRANSB) {
      for (int e = 0; e < 8; ++e) {          // B^T: 128 rows x 32 k
        int g = e * 128 + t, n = g >> 3, c = (g & 7) * 4;
        bR[e] = *(const v4f*)&B[(size_t)(bn + n) * K + kt + c];
      }
    } else {
      for (int e = 0; e < 8; ++e) {          // B: 32 k x 128 n (coalesced along n)
        int g = e * 128 + t, kk = g >> 5, n = (g & 31) * 4;
        bR[e] = *(const v4f*)&B[(size_t)(kt + kk) * N + bn + n];
      }
    }
  };
  auto store_tiles = [&]() {
    for (int e = 0; e < 4; ++e) {
      int g = e * 128 + t, r = g >> 3, c = (g & 7) * 4;
      v4h hv;
      for (int i = 0; i < 4; ++i) hv[i] = (_Float16)aR[e][i];
      *(v4h*)&As[r * LDA + c] = hv;
    }
    if (TRANSB) {
      for (int e = 0; e < 8; ++e) {
        int g = e * 128 + t, n = g >> 3, c = (g & 7) * 4;
        v4h hv;
        for (int i = 0; i < 4; ++i) hv[i] = (_Float16)bR[e][i];
        *(v4h*)&Bs[n * LDA + c] = hv;
      }
    } else {
      for (int e = 0; e < 8; ++e) {          // transpose into Bs[n][k]
        int g = e * 128 + t, kk = g >> 5, n = (g & 31) * 4;
        for (int i = 0; i < 4; ++i) Bs[(n + i) * LDA + kk] = (_Float16)bR[e][i];
      }
    }
  };

  load_tiles(0);
  for (int kt = 0; kt < K; kt += 32) {
    store_tiles();
    __syncthreads();
    if (kt + 32 < K) load_tiles(kt + 32);    // overlap next-tile loads with WMMA
    v16h af[2], bf[4];
    af[0] = load_afrag(As, wm, lane);
    af[1] = load_afrag(As, wm + 16, lane);
    for (int ni = 0; ni < 4; ++ni) bf[ni] = load_bfrag(Bs, wn + ni * 16, lane);
    for (int mi = 0; mi < 2; ++mi)
      for (int ni = 0; ni < 4; ++ni)
        acc[mi][ni] = __builtin_amdgcn_wmma_f32_16x16x32_f16(
            false, af[mi], false, bf[ni], (short)0, acc[mi][ni], false, false);
    __syncthreads();
  }

  // Epilogue: C/D layout -> lane gives N, VGPR r gives M (lanes>=16: M+8)
  const int ncol = lane & 15, mrow = (lane >> 4) * 8;
  for (int mi = 0; mi < 2; ++mi)
    for (int ni = 0; ni < 4; ++ni) {
      int gn = bn + wn + ni * 16 + ncol;
      float bv = bias ? bias[gn] : 0.0f;
      for (int r = 0; r < 8; ++r) {
        int gm = bm + wm + mi * 16 + mrow + r;
        float v = acc[mi][ni][r] + bv;
        if (RELU) v = fmaxf(v, 0.0f);
        C[(size_t)gm * N + gn] = v;
      }
    }
}

// ---------------------------------------------------------------------------
// GAT attention, pass 1: per (row i) branchless online masked softmax stats.
// mask = (Amask[i,j] != 0) || (i == j)   (self-loops)
// Sentinel -1e30 keeps the s=0 accumulator exact with no EXEC divergence.
// ---------------------------------------------------------------------------
__global__ void att_stats(const float* __restrict__ Amask, const float* __restrict__ srcv,
                          const float* __restrict__ dstv, float* __restrict__ mstat,
                          float* __restrict__ dinv, int N, int H) {
  const int i = blockIdx.x, t = threadIdx.x;
  float m[4], s[4], dh[4];
  for (int h = 0; h < H; ++h) { m[h] = -1e30f; s[h] = 0.0f; dh[h] = dstv[i * H + h]; }
  const float* row = Amask + (size_t)i * N;
  for (int j = t; j < N; j += 256) {
    float a = row[j];
    bool msk = (a != 0.0f) || (j == i);
    for (int h = 0; h < H; ++h) {
      float l = dh[h] + srcv[j * H + h];
      l = l >= 0.0f ? l : 0.2f * l;  // leaky_relu 0.2
      float lm = msk ? l : -1e30f;
      float mn = fmaxf(m[h], lm);
      s[h] = s[h] * __expf(m[h] - mn) + (msk ? __expf(l - mn) : 0.0f);
      m[h] = mn;
    }
  }
  __shared__ float sm[4][256], ss[4][256];
  for (int h = 0; h < H; ++h) { sm[h][t] = m[h]; ss[h][t] = s[h]; }
  __syncthreads();
  for (int off = 128; off > 0; off >>= 1) {
    if (t < off)
      for (int h = 0; h < H; ++h) {
        float m1 = sm[h][t], m2 = sm[h][t + off];
        float mm = fmaxf(m1, m2);
        ss[h][t] = ss[h][t] * __expf(m1 - mm) + ss[h][t + off] * __expf(m2 - mm);
        sm[h][t] = mm;
      }
    __syncthreads();
  }
  if (t == 0)
    for (int h = 0; h < H; ++h) {
      mstat[i * H + h] = sm[h][0];
      dinv[i * H + h] = 1.0f / ss[h][0];
    }
}

// ---------------------------------------------------------------------------
// GAT attention, pass 2 (flash-style): out[:,h,:] = relu(P_h @ Xp_h + bias_h)
// P tiles (64x32) built branchlessly on the fly in LDS as f16, fed to WMMA.
// Grid: (F/64, N/64, H). Requires N%32==0, F%64==0.
// ---------------------------------------------------------------------------
__global__ void __launch_bounds__(128)
att_agg(const float* __restrict__ Amask, const float* __restrict__ Xp,
        const float* __restrict__ srcv, const float* __restrict__ dstv,
        const float* __restrict__ mstat, const float* __restrict__ dinv,
        const float* __restrict__ bias, float* __restrict__ out,
        int N, int H, int F) {
  __shared__ _Float16 Ps[64 * LDA];
  __shared__ _Float16 Bs[64 * LDA];
  const int t = threadIdx.x, lane = t & 31, wave = t >> 5;
  const int bm = blockIdx.y * 64;   // node rows
  const int bf = blockIdx.x * 64;   // feature cols within head
  const int h = blockIdx.z;
  const int HF = H * F;
  const int wm = (wave >> 1) * 32, wn = (wave & 1) * 32;
  v8f acc[2][2] = {};

  for (int kt = 0; kt < N; kt += 32) {
    // P tile 64x32: p_ij = mask ? exp(leaky(dst_i+src_j) - m_i) * dinv_i : 0
    for (int e = 0; e < 4; ++e) {
      int g = e * 128 + t, r = g >> 3, c = (g & 7) * 4;
      int gi = bm + r, gj = kt + c;
      v4f av = *(const v4f*)&Amask[(size_t)gi * N + gj];
      float dh = dstv[gi * H + h], mm = mstat[gi * H + h], dd = dinv[gi * H + h];
      // unconditional loads + cndmask select: no EXEC divergence
      float sv[4];
      for (int i = 0; i < 4; ++i) sv[i] = srcv[(gj + i) * H + h];
      v4h pv;
      for (int i = 0; i < 4; ++i) {
        float l = dh + sv[i];
        l = l >= 0.0f ? l : 0.2f * l;
        float pe = __expf(l - mm) * dd;
        bool msk = (av[i] != 0.0f) || (gi == gj + i);
        pv[i] = (_Float16)(msk ? pe : 0.0f);
      }
      *(v4h*)&Ps[r * LDA + c] = pv;
    }
    // Xp tile (32 nodes x 64 feats) staged transposed: Bs[f][j]
    for (int e = 0; e < 4; ++e) {
      int g = e * 128 + t, j = g >> 4, f4 = (g & 15) * 4;
      v4f xv = *(const v4f*)&Xp[(size_t)(kt + j) * HF + h * F + bf + f4];
      for (int i = 0; i < 4; ++i) Bs[(f4 + i) * LDA + j] = (_Float16)xv[i];
    }
    __syncthreads();

    v16h af[2], bfv[2];
    af[0] = load_afrag(Ps, wm, lane);
    af[1] = load_afrag(Ps, wm + 16, lane);
    bfv[0] = load_bfrag(Bs, wn, lane);
    bfv[1] = load_bfrag(Bs, wn + 16, lane);
    for (int mi = 0; mi < 2; ++mi)
      for (int ni = 0; ni < 2; ++ni)
        acc[mi][ni] = __builtin_amdgcn_wmma_f32_16x16x32_f16(
            false, af[mi], false, bfv[ni], (short)0, acc[mi][ni], false, false);
    __syncthreads();
  }

  const int ncol = lane & 15, mrow = (lane >> 4) * 8;
  for (int mi = 0; mi < 2; ++mi)
    for (int ni = 0; ni < 2; ++ni) {
      int gf = bf + wn + ni * 16 + ncol;
      float bv = bias[h * F + gf];
      for (int r = 0; r < 8; ++r) {
        int gm = bm + wm + mi * 16 + mrow + r;
        float v = fmaxf(acc[mi][ni][r] + bv, 0.0f);
        out[(size_t)gm * HF + h * F + gf] = v;
      }
    }
}

// --------------------------- small helper kernels ---------------------------
__global__ void add_eye(const float* __restrict__ A, float* __restrict__ A0, int N) {
  size_t p = (size_t)blockIdx.x * 256 + threadIdx.x;
  size_t i = p / N, j = p % N;
  A0[p] = A[p] + (i == j ? 1.0f : 0.0f);
}

__global__ void gat_scores(const float* __restrict__ Xp, const float* __restrict__ a_src,
                           const float* __restrict__ a_dst, float* __restrict__ srcv,
                           float* __restrict__ dstv, int F, int H) {
  const int i = blockIdx.x, h = blockIdx.y, t = threadIdx.x;
  float s1 = 0.0f, s2 = 0.0f;
  const float* xr = Xp + (size_t)i * H * F + h * F;
  for (int f = t; f < F; f += 128) { float x = xr[f]; s1 += x * a_src[h * F + f]; s2 += x * a_dst[h * F + f]; }
  __shared__ float r1[128], r2[128];
  r1[t] = s1; r2[t] = s2; __syncthreads();
  for (int off = 64; off > 0; off >>= 1) {
    if (t < off) { r1[t] += r1[t + off]; r2[t] += r2[t + off]; }
    __syncthreads();
  }
  if (t == 0) { srcv[i * H + h] = r1[0]; dstv[i * H + h] = r2[0]; }
}

__global__ void pool_score(const float* __restrict__ X, const float* __restrict__ w,
                           const float* __restrict__ b, float* __restrict__ s, int D) {
  const int i = blockIdx.x, t = threadIdx.x;
  float acc = 0.0f;
  for (int f = t; f < D; f += 256) acc += X[(size_t)i * D + f] * w[f];
  __shared__ float r[256];
  r[t] = acc; __syncthreads();
  for (int off = 128; off > 0; off >>= 1) { if (t < off) r[t] += r[t + off]; __syncthreads(); }
  if (t == 0) { float x = r[0] + b[0]; s[i] = 1.0f / (1.0f + __expf(-x)); }
}

// Stable descending rank -> matches jax.lax.top_k ordering (ties by lower index)
__global__ void topk_rank(const float* __restrict__ s, int* __restrict__ idx,
                          float* __restrict__ v, int N, int k) {
  const int i = blockIdx.x, t = threadIdx.x;
  float si = s[i];
  int cnt = 0;
  for (int j = t; j < N; j += 256) {
    float sj = s[j];
    cnt += (sj > si) || (sj == si && j < i);
  }
  __shared__ int r[256];
  r[t] = cnt; __syncthreads();
  for (int off = 128; off > 0; off >>= 1) { if (t < off) r[t] += r[t + off]; __syncthreads(); }
  if (t == 0) { int rank = r[0]; if (rank < k) { idx[rank] = i; v[rank] = si; } }
}

__global__ void gather_adj(const float* __restrict__ Ain, const int* __restrict__ idx,
                           float* __restrict__ G, int N, int k) {
  const int r = blockIdx.x;
  const int ir = idx[r];
  for (int c = threadIdx.x; c < k; c += 256)
    G[(size_t)r * k + c] = Ain[(size_t)ir * N + idx[c]];
}

__global__ void rowsum_rsqrt(const float* __restrict__ G, float* __restrict__ scale, int k) {
  const int r = blockIdx.x, t = threadIdx.x;
  float acc = 0.0f;
  for (int c = t; c < k; c += 256) acc += G[(size_t)r * k + c];
  __shared__ float red[256];
  red[t] = acc; __syncthreads();
  for (int off = 128; off > 0; off >>= 1) { if (t < off) red[t] += red[t + off]; __syncthreads(); }
  if (t == 0) scale[r] = rsqrtf(red[0] + 1e-5f);
}

__global__ void scale_sym(float* __restrict__ G, const float* __restrict__ scale, int k) {
  const int r = blockIdx.x;
  float sr = scale[r];
  for (int c = threadIdx.x; c < k; c += 256) G[(size_t)r * k + c] *= sr * scale[c];
}

__global__ void gather_rows_scale(const float* __restrict__ Xin, const int* __restrict__ idx,
                                  const float* __restrict__ v, float* __restrict__ out, int C) {
  const int r = blockIdx.x;
  const int ir = idx[r];
  float vr = v[r];
  for (int c = threadIdx.x; c < C; c += 256)
    out[(size_t)r * C + c] = Xin[(size_t)ir * C + c] * vr;
}

__global__ void fill_zero(float* __restrict__ p, size_t n) {
  size_t i = (size_t)blockIdx.x * 256 + threadIdx.x;
  if (i < n) p[i] = 0.0f;
}

__global__ void scatter_rows(const float* __restrict__ Xin, const int* __restrict__ idx,
                             float* __restrict__ out, int C) {
  const int r = blockIdx.x;
  const int orow = idx[r];
  for (int c = threadIdx.x; c < C; c += 256)
    out[(size_t)orow * C + c] = Xin[(size_t)r * C + c];
}

__global__ void softmax256(float* __restrict__ Y) {  // in-place row softmax, 256 cols
  const int r = blockIdx.x, t = threadIdx.x;
  float x = Y[(size_t)r * 256 + t];
  __shared__ float red[256];
  red[t] = x; __syncthreads();
  for (int off = 128; off > 0; off >>= 1) { if (t < off) red[t] = fmaxf(red[t], red[t + off]); __syncthreads(); }
  float mx = red[0];
  __syncthreads();
  float e = __expf(x - mx);
  red[t] = e; __syncthreads();
  for (int off = 128; off > 0; off >>= 1) { if (t < off) red[t] += red[t + off]; __syncthreads(); }
  Y[(size_t)r * 256 + t] = e / red[0];
}

// ---------------------------------------------------------------------------
extern "C" void kernel_launch(void* const* d_in, const int* in_sizes, int n_in,
                              void* d_out, int out_size, void* d_ws, size_t ws_size,
                              hipStream_t stream) {
  (void)in_sizes; (void)n_in; (void)out_size; (void)ws_size;
  const float* A    = (const float*)d_in[0];
  const float* X    = (const float*)d_in[1];
  const float* Wd0  = (const float*)d_in[2];
  const float* asd0 = (const float*)d_in[3];
  const float* add0 = (const float*)d_in[4];
  const float* bd0  = (const float*)d_in[5];
  const float* Wd1  = (const float*)d_in[6];
  const float* asd1 = (const float*)d_in[7];
  const float* add1 = (const float*)d_in[8];
  const float* bd1  = (const float*)d_in[9];
  const float* wp0  = (const float*)d_in[10];
  const float* bp0  = (const float*)d_in[11];
  const float* wp1  = (const float*)d_in[12];
  const float* bp1  = (const float*)d_in[13];
  const float* Wb   = (const float*)d_in[14];
  const float* asb  = (const float*)d_in[15];
  const float* adb  = (const float*)d_in[16];
  const float* bb   = (const float*)d_in[17];
  const float* Wu0  = (const float*)d_in[18];
  const float* asu0 = (const float*)d_in[19];
  const float* adu0 = (const float*)d_in[20];
  const float* bu0  = (const float*)d_in[21];
  const float* Wu1  = (const float*)d_in[22];
  const float* asu1 = (const float*)d_in[23];
  const float* adu1 = (const float*)d_in[24];
  const float* bu1  = (const float*)d_in[25];
  const float* Wup  = (const float*)d_in[26];
  // d_in[27] (bup): per-row additive constant cancels in row softmax -> unused.

  float* outAup = (float*)d_out;                   // 8192*8192
  float* outA0  = outAup + 67108864ULL;            // 4096*4096
  float* outA1  = outA0  + 16777216ULL;            // 2048*2048
  float* outR0  = outA1  + 4194304ULL;             // 2048*2048
  float* outR1  = outR0  + 4194304ULL;             // 4096*4096

  float* w = (float*)d_ws;
  size_t o = 0;
  auto alloc = [&](size_t n) -> float* { float* p = w + o; o += (n + 255) & ~(size_t)255; return p; };
  float* Xp0  = alloc(4096 * 512);
  float* X0   = alloc(4096 * 512);
  float* Xp1  = alloc(2048 * 1024);
  float* X1   = alloc(2048 * 1024);
  float* Xpb  = alloc(1024 * 1024);
  float* Xb   = alloc(1024 * 1024);
  float* A2   = alloc(1024 * 1024);
  float* X1in = alloc(2048 * 512);
  float* X2in = alloc(1024 * 1024);
  float* h0   = alloc(2048 * 1024);
  float* Xpu0 = alloc(2048 * 512);
  float* Xu0  = alloc(2048 * 512);
  float* h1   = alloc(4096 * 512);
  float* Xpu1 = alloc(4096 * 256);
  float* Xu1  = alloc(4096 * 256);
  float* Zb   = alloc(8192 * 256);
  float* srcv = alloc(4096 * 4);
  float* dstv = alloc(4096 * 4);
  float* mst  = alloc(4096 * 4);
  float* dnv  = alloc(4096 * 4);
  float* sp   = alloc(8192);
  float* scl  = alloc(4096);
  float* v0   = alloc(2048);
  float* v1   = alloc(1024);
  int* idx0 = (int*)alloc(2048);
  int* idx1 = (int*)alloc(1024);

  // ---- A0 = A + I ----
  add_eye<<<16777216 / 256, 256, 0, stream>>>(A, outA0, 4096);

  // ---- down level 0: GAT(256 -> 4x128) ----
  gemm_wmma<false, false><<<dim3(512 / 128, 4096 / 64), 128, 0, stream>>>(X, Wd0, nullptr, Xp0, 4096, 512, 256);
  gat_scores<<<dim3(4096, 4), 128, 0, stream>>>(Xp0, asd0, add0, srcv, dstv, 128, 4);
  att_stats<<<4096, 256, 0, stream>>>(outA0, srcv, dstv, mst, dnv, 4096, 4);
  att_agg<<<dim3(128 / 64, 4096 / 64, 4), 128, 0, stream>>>(outA0, Xp0, srcv, dstv, mst, dnv, bd0, X0, 4096, 4, 128);

  // ---- pool 0: top-k 2048, symnorm ----
  pool_score<<<4096, 256, 0, stream>>>(X0, wp0, bp0, sp, 512);
  topk_rank<<<4096, 256, 0, stream>>>(sp, idx0, v0, 4096, 2048);
  gather_adj<<<2048, 256, 0, stream>>>(outA0, idx0, outA1, 4096, 2048);
  rowsum_rsqrt<<<2048, 256, 0, stream>>>(outA1, scl, 2048);
  scale_sym<<<2048, 256, 0, stream>>>(outA1, scl, 2048);
  gather_rows_scale<<<2048, 256, 0, stream>>>(X0, idx0, v0, X1in, 512);

  // ---- down level 1: GAT(512 -> 4x256) ----
  gemm_wmma<false, false><<<dim3(1024 / 128, 2048 / 64), 128, 0, stream>>>(X1in, Wd1, nullptr, Xp1, 2048, 1024, 512);
  gat_scores<<<dim3(2048, 4), 128, 0, stream>>>(Xp1, asd1, add1, srcv, dstv, 256, 4);
  att_stats<<<2048, 256, 0, stream>>>(outA1, srcv, dstv, mst, dnv, 2048, 4);
  att_agg<<<dim3(256 / 64, 2048 / 64, 4), 128, 0, stream>>>(outA1, Xp1, srcv, dstv, mst, dnv, bd1, X1, 2048, 4, 256);

  // ---- pool 1: top-k 1024, symnorm ----
  pool_score<<<2048, 256, 0, stream>>>(X1, wp1, bp1, sp, 1024);
  topk_rank<<<2048, 256, 0, stream>>>(sp, idx1, v1, 2048, 1024);
  gather_adj<<<1024, 256, 0, stream>>>(outA1, idx1, A2, 2048, 1024);
  rowsum_rsqrt<<<1024, 256, 0, stream>>>(A2, scl, 1024);
  scale_sym<<<1024, 256, 0, stream>>>(A2, scl, 1024);
  gather_rows_scale<<<1024, 256, 0, stream>>>(X1, idx1, v1, X2in, 1024);

  // ---- bottom: GAT(1024 -> 2x512) ----
  gemm_wmma<false, false><<<dim3(1024 / 128, 1024 / 64), 128, 0, stream>>>(X2in, Wb, nullptr, Xpb, 1024, 1024, 1024);
  gat_scores<<<dim3(1024, 2), 128, 0, stream>>>(Xpb, asb, adb, srcv, dstv, 512, 2);
  att_stats<<<1024, 256, 0, stream>>>(A2, srcv, dstv, mst, dnv, 1024, 2);
  att_agg<<<dim3(512 / 64, 1024 / 64, 2), 128, 0, stream>>>(A2, Xpb, srcv, dstv, mst, dnv, bb, Xb, 1024, 2, 512);

  // ---- up level 0: unpool to 2048, GAT(1024 -> 4x128), rec0 ----
  fill_zero<<<(2048 * 1024) / 256, 256, 0, stream>>>(h0, (size_t)2048 * 1024);
  scatter_rows<<<1024, 256, 0, stream>>>(Xb, idx1, h0, 1024);
  gemm_wmma<false, false><<<dim3(512 / 128, 2048 / 64), 128, 0, stream>>>(h0, Wu0, nullptr, Xpu0, 2048, 512, 1024);
  gat_scores<<<dim3(2048, 4), 128, 0, stream>>>(Xpu0, asu0, adu0, srcv, dstv, 128, 4);
  att_stats<<<2048, 256, 0, stream>>>(outA1, srcv, dstv, mst, dnv, 2048, 4);
  att_agg<<<dim3(128 / 64, 2048 / 64, 4), 128, 0, stream>>>(outA1, Xpu0, srcv, dstv, mst, dnv, bu0, Xu0, 2048, 4, 128);
  gemm_wmma<true, true><<<dim3(2048 / 128, 2048 / 64), 128, 0, stream>>>(Xu0, Xu0, nullptr, outR0, 2048, 2048, 512);

  // ---- up level 1: unpool to 4096, GAT(512 -> 4x64), rec1 ----
  fill_zero<<<(4096 * 512) / 256, 256, 0, stream>>>(h1, (size_t)4096 * 512);
  scatter_rows<<<2048, 256, 0, stream>>>(Xu0, idx0, h1, 512);
  gemm_wmma<false, false><<<dim3(256 / 128, 4096 / 64), 128, 0, stream>>>(h1, Wu1, nullptr, Xpu1, 4096, 256, 512);
  gat_scores<<<dim3(4096, 4), 128, 0, stream>>>(Xpu1, asu1, adu1, srcv, dstv, 64, 4);
  att_stats<<<4096, 256, 0, stream>>>(outA0, srcv, dstv, mst, dnv, 4096, 4);
  att_agg<<<dim3(64 / 64, 4096 / 64, 4), 128, 0, stream>>>(outA0, Xpu1, srcv, dstv, mst, dnv, bu1, Xu1, 4096, 4, 64);
  gemm_wmma<true, true><<<dim3(4096 / 128, 4096 / 64), 128, 0, stream>>>(Xu1, Xu1, nullptr, outR1, 4096, 4096, 256);

  // ---- upsampler: Z = softmax(Wup @ Xu1), A_up = relu(Z @ Z^T) ----
  gemm_wmma<false, false><<<dim3(256 / 128, 8192 / 64), 128, 0, stream>>>(Wup, Xu1, nullptr, Zb, 8192, 256, 4096);
  softmax256<<<8192, 256, 0, stream>>>(Zb);
  gemm_wmma<true, true><<<dim3(8192 / 128, 8192 / 64), 128, 0, stream>>>(Zb, Zb, nullptr, outAup, 8192, 8192, 256);
}
